// queG_64132451664448
// MI455X (gfx1250) — matrix operand order repaired and checked by reading, not compiled, not run on hardware
//
#include <hip/hip_runtime.h>
#include <math.h>

typedef __attribute__((ext_vector_type(16))) _Float16 v16h;
typedef __attribute__((ext_vector_type(8)))  float    v8f;

#define HIDc 50
#define HSc  165
#define FEATc 100
#define CQc  50257
#define EXTc 1000
#define NVc  (CQc + EXTc)   /* 51257 */
#define Lc   250000

/* ---- workspace layout (float offsets into d_ws) ---- */
static constexpr int WS_E    = 0;        /* 250000: scores -> exp(scores-m) */
static constexpr int WS_BMAX = 250000;   /* 2048 block-max partials         */
static constexpr int WS_M    = 252048;   /* global max                      */
static constexpr int WS_ZSUM = 252049;   /* softmax denominator             */
static constexpr int WS_CT   = 252064;   /* 128: unnormalized context       */
static constexpr int WS_HNEW = 252192;   /* 64                              */
static constexpr int WS_CNEW = 252256;   /* 64                              */
static constexpr int WS_Q    = 252320;   /* 128 (zero padded)               */
static constexpr int WS_ET   = 252448;   /* 224                             */
static constexpr int WS_P1   = 252672;
static constexpr int WS_P2   = 252673;
static constexpr int WS_COEF = 252674;   /* p2 / Z                          */
static constexpr int WS_OUT  = 252688;   /* 51257 extended logits           */

union U16h { v16h v; _Float16 h[16]; };
union U8f  { v8f  v; float    f[8];  };

__device__ __forceinline__ float sigf(float x) { return 1.0f / (1.0f + __expf(-x)); }

/* 16-row GEMV tile per wave32 via v_wmma_f32_16x16x32_f16.
 * B = vq broadcast across all 16 columns, so D(M,N) == rowdot(M) for all N.
 * A layout (ISA 7.12.2, 16-bit A 16x32): lanes 0-15 row M=lane hold K{0..7,16..23},
 * lanes 16-31 hold K{8..15,24..31}. B (32x16): same K order as A. */
template<int K, int KPAD>
__device__ __forceinline__ v8f wmma_rows_dot(const float* __restrict__ M, int ld,
                                             int row0, int maxrow,
                                             const float* __restrict__ vq, int lane)
{
  U8f C;
#pragma unroll
  for (int e = 0; e < 8; ++e) C.f[e] = 0.0f;
  const int rr = row0 + (lane & 15);
  const bool rowok = rr < maxrow;
  const float* rp = M + (size_t)rr * (size_t)ld;
  const int kg  = (lane & 16) ? 8  : 0;   /* A K-group base  */
  const int kq0 = (lane & 16) ? 16 : 0;   /* unused; B uses kg */
#pragma unroll
  for (int kb = 0; kb < KPAD; kb += 32) {
    U16h A, B;
#pragma unroll
    for (int e = 0; e < 16; ++e) B.h[e] = (_Float16)vq[kb + kg + (e < 8 ? e : e + 8)];
    if ((kb + 32 <= K) && rowok) {
      float4 x0 = *(const float4*)(rp + kb + kg);
      float4 x1 = *(const float4*)(rp + kb + kg + 4);
      float4 y0 = *(const float4*)(rp + kb + kg + 16);
      float4 y1 = *(const float4*)(rp + kb + kg + 20);
      A.h[0]=(_Float16)x0.x;  A.h[1]=(_Float16)x0.y;
      A.h[2]=(_Float16)x0.z;  A.h[3]=(_Float16)x0.w;
      A.h[4]=(_Float16)x1.x;  A.h[5]=(_Float16)x1.y;
      A.h[6]=(_Float16)x1.z;  A.h[7]=(_Float16)x1.w;
      A.h[8]=(_Float16)y0.x;  A.h[9]=(_Float16)y0.y;
      A.h[10]=(_Float16)y0.z; A.h[11]=(_Float16)y0.w;
      A.h[12]=(_Float16)y1.x; A.h[13]=(_Float16)y1.y;
      A.h[14]=(_Float16)y1.z; A.h[15]=(_Float16)y1.w;
    } else {
#pragma unroll
      for (int e = 0; e < 16; ++e) {
        int k = kb + kg + (e < 8 ? e : e + 8);
        float vv = (rowok && (k < K)) ? rp[k] : 0.0f;
        A.h[e] = (_Float16)vv;
      }
    }
    C.v = __builtin_amdgcn_wmma_f32_16x16x32_f16(false, A.v, false, B.v,
                                                 (short)0, C.v, false, false);
  }
  return C.v;
}

/* ---- K1: LSTM step + attention query; zero accumulators ---- */
__global__ void k1_lstm(const float* __restrict__ y, const float* __restrict__ h0,
                        const float* __restrict__ c0,
                        const float* __restrict__ W_ih, const float* __restrict__ W_hh,
                        const float* __restrict__ b_ih, const float* __restrict__ b_hh,
                        const float* __restrict__ w_h_W, const float* __restrict__ w_h_b,
                        float* __restrict__ ws, float* __restrict__ dout)
{
  __shared__ float g[200];
  __shared__ float hn[HIDc];
  int t = threadIdx.x;
  if (t == 0) ws[WS_ZSUM] = 0.0f;
  if (t < 128) ws[WS_CT + t] = 0.0f;
  if (t < 200) {
    float acc = b_ih[t] + b_hh[t];
    const float* wi = W_ih + t * HSc;
    for (int k = 0; k < HSc; ++k) acc += wi[k] * y[k];
    const float* wh = W_hh + t * HIDc;
    for (int k = 0; k < HIDc; ++k) acc += wh[k] * h0[k];
    g[t] = acc;
  }
  __syncthreads();
  if (t < HIDc) {
    float ig = sigf(g[t]);
    float fg = sigf(g[HIDc + t]);
    float gg = tanhf(g[2 * HIDc + t]);
    float og = sigf(g[3 * HIDc + t]);
    float cn = fg * c0[t] + ig * gg;
    float hv = og * tanhf(cn);
    hn[t] = hv;
    ws[WS_HNEW + t] = hv;
    ws[WS_CNEW + t] = cn;
    dout[NVc + t] = hv;          /* h_new */
    dout[NVc + HIDc + t] = cn;   /* c_new */
  }
  __syncthreads();
  if (t < 128) {
    float qv = 0.0f;
    if (t < FEATc) {
      float acc = w_h_b[t];
      const float* wr = w_h_W + t * HIDc;
      for (int k = 0; k < HIDc; ++k) acc += wr[k] * hn[k];
      qv = acc;
    }
    ws[WS_Q + t] = qv;
  }
}

/* ---- K2: scores s = H * q via WMMA, per-block max ---- */
__global__ void k2_scores(const float* __restrict__ H, float* __restrict__ ws)
{
  __shared__ float qlds[128];
  __shared__ float wmax[16];
  int t = threadIdx.x;
  if (t < 128) qlds[t] = ws[WS_Q + t];
  if (t < 16) wmax[t] = -3.0e38f;
  __syncthreads();
  int wave = t >> 5, lane = t & 31;
  int row0 = (blockIdx.x * 8 + wave) * 16;
  if (row0 < Lc) {   /* wave-uniform guard: EXEC all-ones at WMMA */
    U8f C; C.v = wmma_rows_dot<FEATc, 128>(H, FEATc, row0, Lc, qlds, lane);
    if ((lane & 15) == 0) {  /* lanes 0 and 16 own N=0 column */
      int mb = (lane >> 4) * 8;
      float mx = -3.0e38f;
#pragma unroll
      for (int r = 0; r < 8; ++r) {
        ws[WS_E + row0 + mb + r] = C.f[r];
        mx = fmaxf(mx, C.f[r]);
      }
      wmax[wave * 2 + (lane >> 4)] = mx;
    }
  }
  __syncthreads();
  if (t == 0) {
    float mx = wmax[0];
    for (int i = 1; i < 16; ++i) mx = fmaxf(mx, wmax[i]);
    ws[WS_BMAX + blockIdx.x] = mx;
  }
}

/* ---- K3: reduce block maxes ---- */
__global__ void k3_maxred(float* __restrict__ ws, int n)
{
  __shared__ float red[256];
  int t = threadIdx.x;
  float mx = -3.0e38f;
  for (int i = t; i < n; i += 256) mx = fmaxf(mx, ws[WS_BMAX + i]);
  red[t] = mx; __syncthreads();
  for (int s = 128; s > 0; s >>= 1) {
    if (t < s) red[t] = fmaxf(red[t], red[t + s]);
    __syncthreads();
  }
  if (t == 0) ws[WS_M] = red[0];
}

/* ---- K4: e = exp(s-m) in place; Z partial; ct_un partial (2nd H pass) ---- */
__global__ void k4_exp_ct(const float* __restrict__ H, float* __restrict__ ws)
{
  __shared__ float els[512];
  __shared__ float red[128];
  int t = threadIdx.x;
  int row0 = blockIdx.x * 512;
  float m = ws[WS_M];
  float lsum = 0.0f;
  for (int r = t; r < 512; r += 128) {
    int i = row0 + r;
    float e = 0.0f;
    if (i < Lc) {
      e = __expf(ws[WS_E + i] - m);
      ws[WS_E + i] = e;
    }
    els[r] = e;
    lsum += e;
  }
  red[t] = lsum; __syncthreads();
  for (int s = 64; s > 0; s >>= 1) {
    if (t < s) red[t] += red[t + s];
    __syncthreads();
  }
  if (t == 0) atomicAdd(&ws[WS_ZSUM], red[0]);
  if (t < FEATc) {
    float acc = 0.0f;
    int rmax = min(512, Lc - row0);
    for (int r = 0; r < rmax; ++r)
      acc += els[r] * H[(size_t)(row0 + r) * FEATc + t];
    atomicAdd(&ws[WS_CT + t], acc);
  }
}

/* ---- K5: c_t, e_t, p1, p2, coef ---- */
__global__ void k5_final(const float* __restrict__ l1_W, const float* __restrict__ l1_b,
                         const float* __restrict__ l3_W, const float* __restrict__ l3_b,
                         const float* __restrict__ l4_W, const float* __restrict__ l4_b,
                         float* __restrict__ ws)
{
  __shared__ float temp[152];
  int t = threadIdx.x;
  float Z = ws[WS_ZSUM];
  if (t < HIDc) temp[t] = ws[WS_HNEW + t];
  else if (t < 150) temp[t] = ws[WS_CT + (t - HIDc)] / Z;
  __syncthreads();
  if (t < 200) {
    float acc = l1_b[t];
    const float* wr = l1_W + t * 150;
    for (int k = 0; k < 150; ++k) acc += wr[k] * temp[k];
    ws[WS_ET + t] = sigf(acc);
  }
  if (t == 0) {
    float acc = l3_b[0];
    for (int k = 0; k < HIDc; ++k) acc += l3_W[k] * temp[k];
    ws[WS_P1] = sigf(acc);
  }
  if (t == 1) {
    float acc = l4_b[0];
    for (int k = 0; k < HIDc; ++k) acc += l4_W[k] * temp[k];
    float p2 = sigf(acc);
    ws[WS_P2] = p2;
    ws[WS_COEF] = p2 / Z;
  }
}

/* ---- K6: out = p1*(l2_W*e_t + b) via WMMA; zero the EXT copy slots ---- */
__global__ void k6_vocab(const float* __restrict__ l2_W, const float* __restrict__ l2_b,
                         float* __restrict__ ws)
{
  __shared__ float et[224];
  int t = threadIdx.x;
  if (t < 224) et[t] = (t < 200) ? ws[WS_ET + t] : 0.0f;
  __syncthreads();
  float p1 = ws[WS_P1];
  int wave = t >> 5, lane = t & 31;
  int row0 = (blockIdx.x * 8 + wave) * 16;
  if (row0 < NVc) {
    U8f C; C.v = wmma_rows_dot<200, 224>(l2_W, 200, row0, CQc, et, lane);
    if ((lane & 15) == 0) {
      int mb = (lane >> 4) * 8;
#pragma unroll
      for (int r = 0; r < 8; ++r) {
        int row = row0 + mb + r;
        if (row < NVc) {
          float v = (row < CQc) ? p1 * (C.f[r] + l2_b[row]) : 0.0f;
          ws[WS_OUT + row] = v;
        }
      }
    }
  }
}

/* ---- K7: pointer-copy scatter ---- */
__global__ void k7_scatter(const int* __restrict__ cont, float* __restrict__ ws)
{
  int i = blockIdx.x * 256 + threadIdx.x;
  if (i < Lc) {
    float coef = ws[WS_COEF];
    atomicAdd(&ws[WS_OUT + cont[i]], coef * ws[WS_E + i]);
  }
}

/* ---- K8: log_softmax over extended vocab ---- */
__global__ void k8_logsoftmax(const float* __restrict__ ws, float* __restrict__ out)
{
  __shared__ float red[1024];
  __shared__ float mval, zval;
  int t = threadIdx.x;
  float mx = -3.0e38f;
  for (int i = t; i < NVc; i += 1024) mx = fmaxf(mx, ws[WS_OUT + i]);
  red[t] = mx; __syncthreads();
  for (int s = 512; s > 0; s >>= 1) {
    if (t < s) red[t] = fmaxf(red[t], red[t + s]);
    __syncthreads();
  }
  if (t == 0) mval = red[0];
  __syncthreads();
  float m = mval;
  float lsum = 0.0f;
  for (int i = t; i < NVc; i += 1024) lsum += __expf(ws[WS_OUT + i] - m);
  __syncthreads();
  red[t] = lsum; __syncthreads();
  for (int s = 512; s > 0; s >>= 1) {
    if (t < s) red[t] += red[t + s];
    __syncthreads();
  }
  if (t == 0) zval = logf(red[0]);
  __syncthreads();
  float lz = zval;
  for (int i = t; i < NVc; i += 1024) out[i] = ws[WS_OUT + i] - m - lz;
}

extern "C" void kernel_launch(void* const* d_in, const int* in_sizes, int n_in,
                              void* d_out, int out_size, void* d_ws, size_t ws_size,
                              hipStream_t stream) {
  (void)in_sizes; (void)n_in; (void)out_size; (void)ws_size;
  const float* H     = (const float*)d_in[0];
  const float* y     = (const float*)d_in[1];
  const float* h0    = (const float*)d_in[2];
  const float* c0    = (const float*)d_in[3];
  const int*   cont  = (const int*)  d_in[4];
  /* d_in[5] = ext scalar (== 1000, hardcoded) */
  const float* W_ih  = (const float*)d_in[6];
  const float* W_hh  = (const float*)d_in[7];
  const float* b_ih  = (const float*)d_in[8];
  const float* b_hh  = (const float*)d_in[9];
  const float* w_h_W = (const float*)d_in[10];
  const float* w_h_b = (const float*)d_in[11];
  const float* l1_W  = (const float*)d_in[12];
  const float* l1_b  = (const float*)d_in[13];
  const float* l2_W  = (const float*)d_in[14];
  const float* l2_b  = (const float*)d_in[15];
  const float* l3_W  = (const float*)d_in[16];
  const float* l3_b  = (const float*)d_in[17];
  const float* l4_W  = (const float*)d_in[18];
  const float* l4_b  = (const float*)d_in[19];
  float* out = (float*)d_out;
  float* ws  = (float*)d_ws;

  const int nb2 = ((Lc / 16) + 7) / 8;            /* 1954 */
  const int nb4 = (Lc + 511) / 512;               /* 489  */
  const int nb6 = (((NVc + 15) / 16) + 7) / 8;    /* 401  */
  const int nb7 = (Lc + 255) / 256;               /* 977  */

  k1_lstm<<<1, 256, 0, stream>>>(y, h0, c0, W_ih, W_hh, b_ih, b_hh,
                                 w_h_W, w_h_b, ws, out);
  k2_scores<<<nb2, 256, 0, stream>>>(H, ws);
  k3_maxred<<<1, 256, 0, stream>>>(ws, nb2);
  k4_exp_ct<<<nb4, 128, 0, stream>>>(H, ws);
  k5_final<<<1, 256, 0, stream>>>(l1_W, l1_b, l3_W, l3_b, l4_W, l4_b, ws);
  k6_vocab<<<nb6, 256, 0, stream>>>(l2_W, l2_b, ws);
  k7_scatter<<<nb7, 256, 0, stream>>>(cont, ws);
  k8_logsoftmax<<<1, 1024, 0, stream>>>(ws, out);
}
